// CoarseEncoder_15470472200216
// MI455X (gfx1250) — matrix operand order repaired and checked by reading, not compiled, not run on hardware
//
#include <hip/hip_runtime.h>
#include <math.h>

typedef __attribute__((ext_vector_type(16))) _Float16 v16h;
typedef __attribute__((ext_vector_type(8)))  _Float16 v8h;
typedef __attribute__((ext_vector_type(8)))  float    v8f;

#define NPTS   262144
#define NSCN   64
#define PTS    4096          // points per scene
#define KNBR   64
#define CIN    256
#define CMID   256
#define COUT   512
#define KPAD   288           // 259 padded to 9*32
#define KC1    9             // K-chunks layer 1
#define KC2    8             // K-chunks layer 2 / output

// ---------------------------------------------------------------- kernel 1
__global__ void k_centroid(const float* __restrict__ pos, float* __restrict__ centers) {
    __shared__ float sx[256], sy[256], sz[256];
    const int b = blockIdx.x, t = threadIdx.x;
    const int base = b * PTS;
    float x = 0.f, y = 0.f, z = 0.f;
    for (int j = t; j < PTS; j += 256) {
        const float* p = pos + (size_t)(base + j) * 3;
        x += p[0]; y += p[1]; z += p[2];
    }
    sx[t] = x; sy[t] = y; sz[t] = z;
    __syncthreads();
    for (int s = 128; s > 0; s >>= 1) {
        if (t < s) { sx[t] += sx[t + s]; sy[t] += sy[t + s]; sz[t] += sz[t + s]; }
        __syncthreads();
    }
    if (t == 0) {
        const float inv = 1.0f / (float)PTS;   // counts == 4096 for every scene
        centers[b * 3 + 0] = sx[0] * inv;
        centers[b * 3 + 1] = sy[0] * inv;
        centers[b * 3 + 2] = sz[0] * inv;
    }
}

// ---------------------------------------------------------------- kernel 2
static __device__ inline unsigned long long umin64(unsigned long long a, unsigned long long b) {
    return a < b ? a : b;
}

__global__ void k_topk(const float* __restrict__ pos, const float* __restrict__ centers,
                       int* __restrict__ nbrIdx, float* __restrict__ validMask,
                       float* __restrict__ validCount) {
    __shared__ float d2s[PTS];
    __shared__ unsigned long long red[256];
    const int b = blockIdx.x, t = threadIdx.x;
    const int base = b * PTS;
    const float cx = centers[b * 3 + 0], cy = centers[b * 3 + 1], cz = centers[b * 3 + 2];
    for (int j = t; j < PTS; j += 256) {
        const float* p = pos + (size_t)(base + j) * 3;
        const float dx = p[0] - cx, dy = p[1] - cy, dz = p[2] - cz;
        d2s[j] = dx * dx + dy * dy + dz * dz;
    }
    __syncthreads();
    for (int k = 0; k < KNBR; ++k) {
        unsigned long long best = ~0ull;
        const int j0 = t * (PTS / 256);
        #pragma unroll 4
        for (int q = 0; q < PTS / 256; ++q) {
            const int j = j0 + q;
            const unsigned long long key =
                ((unsigned long long)__float_as_uint(d2s[j]) << 32) | (unsigned int)j;
            best = umin64(best, key);
        }
        red[t] = best;
        __syncthreads();
        for (int s = 128; s > 0; s >>= 1) {
            if (t < s) red[t] = umin64(red[t], red[t + s]);
            __syncthreads();
        }
        if (t == 0) {
            const unsigned long long w = red[0];
            const int j = (int)(w & 0xffffffffu);
            const float dv = __uint_as_float((unsigned int)(w >> 32));
            nbrIdx[b * KNBR + k]    = base + j;
            validMask[b * KNBR + k] = (dv <= 100.0f) ? 1.0f : 0.0f;  // RADIUS^2
            d2s[j] = __uint_as_float(0x7f800000u);                   // +inf, remove from pool
        }
        __syncthreads();
    }
    if (t == 0) {
        float c = 0.f;
        for (int k = 0; k < KNBR; ++k) c += validMask[b * KNBR + k];
        validCount[b] = c;
    }
}

// ---------------------------------------------------------------- fragment helpers
// A (16x32 f16): lane L -> M = L&15 ; half h -> K = (h<8 ? h : h+8) + 8*(L>=16)
//   => two contiguous 16B runs at K = 8*hi and K = 8*hi+16 in a row-major row.
// B (32x16 f16): lane L -> N = L&15 ; half h -> K = (h<8 ? h : h+8) + 8*(L>=16)
//   => stored pre-swizzled in LDS as [ntile][kc][lane][16 contiguous halves].
// D (16x16 f32): elem r -> M = r + 8*(L>=16), N = L&15.

static __device__ inline v16h load_a_frag(const _Float16* rowBase /*16B aligned*/) {
    v8h lo = *(const v8h*)(rowBase);
    v8h hi = *(const v8h*)(rowBase + 16);
    return __builtin_shufflevector(lo, hi, 0, 1, 2, 3, 4, 5, 6, 7,
                                           8, 9, 10, 11, 12, 13, 14, 15);
}
static __device__ inline v16h load_b_frag(const _Float16* p /*16B aligned*/) {
    v8h lo = *(const v8h*)(p);
    v8h hi = *(const v8h*)(p + 8);
    return __builtin_shufflevector(lo, hi, 0, 1, 2, 3, 4, 5, 6, 7,
                                           8, 9, 10, 11, 12, 13, 14, 15);
}
// destination index in swizzled B buffer for element (K, N), nkc = K-chunks
static __device__ inline int bswz_idx(int K, int N, int nkc) {
    const int kc = K >> 5, kl = K & 31;
    const int h = (kl & 7) + ((kl >> 4) << 3), hib = (kl >> 3) & 1;
    const int lane = (N & 15) + (hib << 4);
    const int nt = N >> 4;
    return (((nt * nkc) + kc) * 32 + lane) * 16 + h;
}

// ---------------------------------------------------------------- kernel 3
__global__ void k_mlp(const float* __restrict__ pos, const float* __restrict__ feature,
                      const float* __restrict__ W1, const float* __restrict__ b1,
                      const float* __restrict__ W2, const float* __restrict__ b2,
                      const float* __restrict__ centers, const int* __restrict__ nbrIdx,
                      const float* __restrict__ validMask, const float* __restrict__ validCount,
                      float* __restrict__ agg) {
    extern __shared__ char smem[];
    _Float16* sW = (_Float16*)smem;            // swizzled weights: 16*KC1*32*16 = 73728 halves
    _Float16* sE = sW + 16 * KC1 * 32 * 16;    // [64][KPAD] row-major; reused as H2 [64][256]
    _Float16* sH = sE + KNBR * KPAD;           // [64][256] row-major

    const int b = blockIdx.x, t = threadIdx.x;
    const int lane = t & 31, wave = t >> 5;
    const int mRow = lane & 15, hi = lane >> 4;
    const float cx = centers[b * 3 + 0], cy = centers[b * 3 + 1], cz = centers[b * 3 + 2];

    // stage W1 swizzled (coalesced global reads, scattered LDS b16 writes)
    for (int i = t; i < (CIN + 3) * 256; i += 256) {
        const int K = i >> 8, N = i & 255;
        sW[bswz_idx(K, N, KC1)] = (_Float16)W1[i];
    }
    for (int i = t; i < (KPAD - (CIN + 3)) * 256; i += 256) {   // zero padding rows
        const int K = (CIN + 3) + (i >> 8), N = i & 255;
        sW[bswz_idx(K, N, KC1)] = (_Float16)0.0f;
    }
    // stage E = concat(feature[idx], pos[idx]-center), zero-padded, row-major
    for (int i = t; i < KNBR * KPAD; i += 256) {
        const int k = i / KPAD, c = i % KPAD;
        const int g = nbrIdx[b * KNBR + k];
        float v;
        if (c < CIN)        v = feature[(size_t)g * CIN + c];
        else if (c == 256)  v = pos[(size_t)g * 3 + 0] - cx;
        else if (c == 257)  v = pos[(size_t)g * 3 + 1] - cy;
        else if (c == 258)  v = pos[(size_t)g * 3 + 2] - cz;
        else                v = 0.0f;
        sE[i] = (_Float16)v;
    }
    __syncthreads();

    // ---- layer 1: H1 = relu(E @ W1 + b1): 4x16 tiles, 9 K-chunks
    for (int tt = wave; tt < 64; tt += 8) {
        const int M0 = (tt & 3) * 16, nt = tt >> 2, N0 = nt * 16;
        const _Float16* aRow  = &sE[(M0 + mRow) * KPAD + (hi << 3)];
        const _Float16* bBase = &sW[(nt * KC1 * 32 + lane) * 16];
        v8f acc = {};
        #pragma unroll
        for (int kc = 0; kc < KC1; ++kc) {
            const v16h av = load_a_frag(aRow + kc * 32);
            const v16h bv = load_b_frag(bBase + kc * 512);
            acc = __builtin_amdgcn_wmma_f32_16x16x32_f16(false, av, false, bv,
                                                         (short)0, acc, false, false);
        }
        #pragma unroll
        for (int r = 0; r < 8; ++r) {
            const int M = M0 + r + (hi << 3), N = N0 + mRow;
            const float v = acc[r] + b1[N];
            sH[M * 256 + N] = (_Float16)(v > 0.f ? v : 0.f);
        }
    }
    __syncthreads();

    // stage W2 swizzled
    for (int i = t; i < 256 * 256; i += 256) {
        const int K = i >> 8, N = i & 255;
        sW[bswz_idx(K, N, KC2)] = (_Float16)W2[i];
    }
    __syncthreads();

    // ---- layer 2: H2 = relu(H1 @ W2 + b2) -> reuse sE as [64][256]
    for (int tt = wave; tt < 64; tt += 8) {
        const int M0 = (tt & 3) * 16, nt = tt >> 2, N0 = nt * 16;
        const _Float16* aRow  = &sH[(M0 + mRow) * 256 + (hi << 3)];
        const _Float16* bBase = &sW[(nt * KC2 * 32 + lane) * 16];
        v8f acc = {};
        #pragma unroll
        for (int kc = 0; kc < KC2; ++kc) {
            const v16h av = load_a_frag(aRow + kc * 32);
            const v16h bv = load_b_frag(bBase + kc * 512);
            acc = __builtin_amdgcn_wmma_f32_16x16x32_f16(false, av, false, bv,
                                                         (short)0, acc, false, false);
        }
        #pragma unroll
        for (int r = 0; r < 8; ++r) {
            const int M = M0 + r + (hi << 3), N = N0 + mRow;
            const float v = acc[r] + b2[N];
            sE[M * 256 + N] = (_Float16)(v > 0.f ? v : 0.f);
        }
    }
    __syncthreads();

    // ---- masked mean over K neighbors (one column per thread)
    float s = 0.f;
    for (int k = 0; k < KNBR; ++k)
        s += (float)sE[k * 256 + t] * validMask[b * KNBR + k];
    agg[b * 256 + t] = s / fmaxf(validCount[b], 1.0f);
}

// ---------------------------------------------------------------- kernel 4
// AGG[64,256] @ W3[256,512] + b3, 4 blocks x 128 output columns each
__global__ void k_out(const float* __restrict__ agg, const float* __restrict__ W3,
                      const float* __restrict__ b3, float* __restrict__ out) {
    extern __shared__ char smem[];
    _Float16* sA = (_Float16*)smem;       // [64][256] row-major     (32 KB)
    _Float16* sB = sA + KNBR * 256;       // swizzled 8nt*8kc*32*16  (64 KB)

    const int nb = blockIdx.x, t = threadIdx.x;
    const int lane = t & 31, wave = t >> 5;
    const int mRow = lane & 15, hi = lane >> 4;

    for (int i = t; i < KNBR * 256; i += 256) sA[i] = (_Float16)agg[i];
    for (int i = t; i < 256 * 128; i += 256) {
        const int K = i >> 7, nl = i & 127;
        sB[bswz_idx(K, nl, KC2)] = (_Float16)W3[K * COUT + nb * 128 + nl];
    }
    __syncthreads();

    for (int tt = wave; tt < 32; tt += 8) {           // 4 M-tiles x 8 local N-tiles
        const int M0 = (tt & 3) * 16, nt = tt >> 2;
        const int N0 = nb * 128 + nt * 16;
        const _Float16* aRow  = &sA[(M0 + mRow) * 256 + (hi << 3)];
        const _Float16* bBase = &sB[(nt * KC2 * 32 + lane) * 16];
        v8f acc = {};
        #pragma unroll
        for (int kc = 0; kc < KC2; ++kc) {
            const v16h av = load_a_frag(aRow + kc * 32);
            const v16h bv = load_b_frag(bBase + kc * 512);
            acc = __builtin_amdgcn_wmma_f32_16x16x32_f16(false, av, false, bv,
                                                         (short)0, acc, false, false);
        }
        #pragma unroll
        for (int r = 0; r < 8; ++r) {
            const int M = M0 + r + (hi << 3), N = N0 + mRow;
            out[M * COUT + N] = acc[r] + b3[N];
        }
    }
}

// ---------------------------------------------------------------- kernel 5
__global__ void k_final(const float* __restrict__ out, const float* __restrict__ eps,
                        float* __restrict__ dst) {
    const int i = blockIdx.x * 256 + threadIdx.x;  // 0..16383
    const int b = i >> 8, c = i & 255;
    const float mu = out[b * COUT + c];
    const float s  = out[b * COUT + 256 + c];
    const float sigma = log1pf(expf(-fabsf(s))) + fmaxf(s, 0.f);  // stable softplus
    dst[i]                  = mu + sigma * eps[i];  // z
    dst[NSCN * 256 + i]     = mu;                   // mu
    dst[2 * NSCN * 256 + i] = sigma;                // sigma
}

// ---------------------------------------------------------------- launch
extern "C" void kernel_launch(void* const* d_in, const int* in_sizes, int n_in,
                              void* d_out, int out_size, void* d_ws, size_t ws_size,
                              hipStream_t stream) {
    const float* pos     = (const float*)d_in[0];
    const float* feature = (const float*)d_in[1];
    /* d_in[2] = batch (contiguous repeat(arange(B), N/B)) — layout known */
    const float* eps     = (const float*)d_in[3];
    const float* W1      = (const float*)d_in[4];
    const float* b1      = (const float*)d_in[5];
    const float* W2      = (const float*)d_in[6];
    const float* b2      = (const float*)d_in[7];
    const float* W3      = (const float*)d_in[8];
    const float* b3      = (const float*)d_in[9];

    char* ws = (char*)d_ws;
    float* centers    = (float*)(ws + 0);                 //   768 B
    int*   nbrIdx     = (int*)  (ws + 1024);              //  16 KB
    float* validMask  = (float*)(ws + 1024 + 16384);      //  16 KB
    float* validCount = (float*)(ws + 1024 + 32768);      //   256 B
    float* agg        = (float*)(ws + 34816);             //  64 KB
    float* outbuf     = (float*)(ws + 34816 + 65536);     // 128 KB

    k_centroid<<<NSCN, 256, 0, stream>>>(pos, centers);
    k_topk    <<<NSCN, 256, 0, stream>>>(pos, centers, nbrIdx, validMask, validCount);

    const size_t lds3 = ((size_t)16 * KC1 * 32 * 16 + (size_t)KNBR * KPAD +
                         (size_t)KNBR * 256) * sizeof(_Float16);
    k_mlp<<<NSCN, 256, lds3, stream>>>(pos, feature, W1, b1, W2, b2,
                                       centers, nbrIdx, validMask, validCount, agg);

    const size_t lds4 = ((size_t)KNBR * 256 + (size_t)8 * KC2 * 32 * 16) * sizeof(_Float16);
    k_out  <<<4, 256, lds4, stream>>>(agg, W3, b3, outbuf);
    k_final<<<64, 256, 0, stream>>>(outbuf, eps, (float*)d_out);
}
